// SigmoidAttention_54589034332986
// MI455X (gfx1250) — compile-verified, matrix operand order checked by reading, not run on hardware
//
#include <hip/hip_runtime.h>
#include <hip/hip_bf16.h>

// ---------------------------------------------------------------------------
// Sigmoid attention for MI455X (gfx1250, wave32, WMMA f16 16x16x32 -> f32)
// B=2, S=4096, D=768, H=12, DH=64
// Async LDS staging (ASYNCcnt) + double buffering + 32-row wave tiles.
// ---------------------------------------------------------------------------

typedef _Float16 f16;
typedef __attribute__((ext_vector_type(16))) _Float16 v16h;
typedef __attribute__((ext_vector_type(8)))  _Float16 v8h;
typedef __attribute__((ext_vector_type(4)))  _Float16 v4h;
typedef __attribute__((ext_vector_type(8)))  float    v8f;

constexpr int Bc  = 2;
constexpr int Sc  = 4096;
constexpr int Dc  = 768;
constexpr int Hc  = 12;
constexpr int DHc = 64;

// Fragment layout for 16-bit A/B operands (CDNA5 ISA 7.12.2, wave32):
// lane L holds row (L & 15); k elements [kh .. kh+7] in elems 0..7 and
// [16+kh .. 16+kh+7] in elems 8..15, where kh = (L>>4)*8.
__device__ __forceinline__ v16h cat8(v8h lo, v8h hi) {
    return __builtin_shufflevector(lo, hi, 0,1,2,3,4,5,6,7,8,9,10,11,12,13,14,15);
}
__device__ __forceinline__ v16h load_frag(const f16* rowptr, int kh) {
    v8h lo = *(const v8h*)(rowptr + kh);
    v8h hi = *(const v8h*)(rowptr + 16 + kh);
    return cat8(lo, hi);
}
__device__ __forceinline__ v8f wmma_f16(v16h a, v16h b, v8f c) {
    return __builtin_amdgcn_wmma_f32_16x16x32_f16(false, a, false, b, (short)0, c, false, false);
}

// Async 16B global->LDS copy (GLOBAL_LOAD_ASYNC_TO_LDS_B128, ASYNCcnt).
// Generic LDS addresses carry the LDS byte offset in the low 32 bits.
__device__ __forceinline__ void async_copy_b128(f16* lds_dst, const f16* src) {
    unsigned int       l = (unsigned int)(unsigned long long)(uintptr_t)lds_dst;
    unsigned long long g = (unsigned long long)(uintptr_t)src;
    asm volatile("global_load_async_to_lds_b128 %0, %1, off"
                 :: "v"(l), "v"(g) : "memory");
}
#define WAIT_ASYNCCNT(imm) asm volatile("s_wait_asynccnt " imm ::: "memory")

// ---------------------------------------------------------------------------
// Pass 0: f32 -> f16 conversion (one-time; all sizes are multiples of 4)
// ---------------------------------------------------------------------------
__global__ __launch_bounds__(256) void cvt_f32_f16(const float* __restrict__ src,
                                                   f16* __restrict__ dst, int n) {
    int idx = (blockIdx.x * blockDim.x + threadIdx.x) * 4;
    if (idx + 3 < n) {
        float4 v = *(const float4*)(src + idx);
        v4h o;
        o[0] = (f16)v.x; o[1] = (f16)v.y; o[2] = (f16)v.z; o[3] = (f16)v.w;
        *(v4h*)(dst + idx) = o;
    } else {
        for (int k = idx; k < n; ++k) dst[k] = (f16)src[k];
    }
}

// ---------------------------------------------------------------------------
// Pass 1: QKV projection  Y = X @ W^T  (M=8192, N=768, K=768), f16 WMMA.
// Per workgroup: 4 waves x 32 rows = 128 rows x 64 cols.
// W k-slab [64 n][32 k] staged in LDS via async loads, double-buffered,
// shared by all 4 waves. Each B fragment feeds 2 WMMAs (2 M-tiles).
// w==0 -> Qh [8192][768], w==1 -> Kh [8192][768], w==2 -> Vt [B][H][DH][S].
// Grid: (8192/128, 768/64, 3), block = 128.
// ---------------------------------------------------------------------------
__global__ __launch_bounds__(128) void proj_wmma(
    const f16* __restrict__ Xh,
    const f16* __restrict__ Wq, const f16* __restrict__ Wk, const f16* __restrict__ Wv,
    f16* __restrict__ Qh, f16* __restrict__ Kh, f16* __restrict__ Vt) {

    __shared__ f16 sW[2][64 * 32];   // [n][k], double-buffered (2 x 4KB)
    __shared__ f16 sO[4][16 * 64];   // per-wave store-retile scratch (4 x 2KB)

    const int tid  = threadIdx.x;
    const int wave = tid >> 5;
    const int lane = tid & 31;
    const int m0   = (blockIdx.x * 4 + wave) * 32;   // flat row over B*S
    const int n0   = blockIdx.y * 64;                // output column tile
    const int w    = blockIdx.z;

    const f16* W = (w == 0) ? Wq : (w == 1) ? Wk : Wv;
    const int kh = (lane >> 4) * 8;

    // stage W[n0..n0+63][kk..kk+31]: 256 x 16B chunks, 2 per thread
    auto issueW = [&](int buf, int kk) {
#pragma unroll
        for (int i = 0; i < 2; ++i) {
            const int c = tid + i * 128;
            const int n = c >> 2, ko = (c & 3) * 8;
            async_copy_b128(&sW[buf][n * 32 + ko],
                            &W[(size_t)(n0 + n) * Dc + kk + ko]);
        }
    };

    v8f c[2][4] = {};
    issueW(0, 0);
    for (int kk = 0; kk < Dc; kk += 32) {
        const int cur = (kk >> 5) & 1;
        issueW(cur ^ 1, (kk + 32 < Dc) ? kk + 32 : 0);   // prefetch next slab
        WAIT_ASYNCCNT("0x2");   // in-order: first 2 (current slab) complete
        __syncthreads();

        v16h a[2];
#pragma unroll
        for (int mt = 0; mt < 2; ++mt)
            a[mt] = load_frag(Xh + (size_t)(m0 + mt * 16 + (lane & 15)) * Dc + kk, kh);
#pragma unroll
        for (int j = 0; j < 4; ++j) {
            v16h bf = load_frag(&sW[cur][(j * 16 + (lane & 15)) * 32], kh);
#pragma unroll
            for (int mt = 0; mt < 2; ++mt)
                c[mt][j] = wmma_f16(a[mt], bf, c[mt][j]);
        }
        __syncthreads();        // all waves done with sW[cur] before overwrite
    }

    if (w < 2) {
        f16* Y = (w == 0) ? Qh : Kh;
        f16* so = &sO[wave][0];
#pragma unroll
        for (int mt = 0; mt < 2; ++mt) {
            // retile D-layout (lane=col) -> row-major [16][64] in LDS
#pragma unroll
            for (int j = 0; j < 4; ++j)
#pragma unroll
                for (int r = 0; r < 8; ++r)
                    so[((lane >> 4) * 8 + r) * 64 + j * 16 + (lane & 15)] = (f16)c[mt][j][r];
            __syncthreads();    // uniform; also orders per-wave DS write->read
            // coalesced b128 stores: 128 chunks, 4 per lane
#pragma unroll
            for (int i = 0; i < 4; ++i) {
                const int ch = lane + i * 32;
                const int row = ch >> 3, co = (ch & 7) * 8;
                *(v8h*)&Y[(size_t)(m0 + mt * 16 + row) * Dc + n0 + co] =
                    *(const v8h*)&so[row * 64 + co];
            }
            __syncthreads();
        }
    } else {
        // Vt[b][h][dh][s], n0 is 64-aligned -> whole tile is one head.
        const int h    = blockIdx.y;            // 768/64 == H
        const int bidx = m0 / Sc;
#pragma unroll
        for (int mt = 0; mt < 2; ++mt) {
            const int srow = (m0 % Sc) + mt * 16 + (lane >> 4) * 8;
#pragma unroll
            for (int j = 0; j < 4; ++j) {
                const int dh = j * 16 + (lane & 15);
                v8h p;
#pragma unroll
                for (int r = 0; r < 8; ++r) p[r] = (f16)c[mt][j][r];
                *(v8h*)(Vt + (((size_t)bidx * Hc + h) * DHc + dh) * Sc + srow) = p;
            }
        }
    }
}

// ---------------------------------------------------------------------------
// Pass 2: sigmoid attention.
// Workgroup = 4 waves x 32 query rows = 128 q-rows per (b,h) block.
// 32-key blocks: K tile [32][64] + V tile [64][32] async-staged into LDS,
// double-buffered, shared by all 4 waves. Per block per wave:
//   8 score WMMAs (2 M-tiles x 2 key-groups x K=64 chained),
//   sigmoid in VALU, P transposed via per-wave LDS scratch,
//   8 accumulation WMMAs (2 M-tiles x 4 dh-tiles).
// Grid: (S/128, H, B), block = 128.
// ---------------------------------------------------------------------------
__global__ __launch_bounds__(128) void attn_wmma(
    const f16* __restrict__ Qh, const f16* __restrict__ Kh,
    const f16* __restrict__ Vt, float* __restrict__ Out) {

    __shared__ f16 sK[2][32 * 64];   // [key][dh]   2 x 4KB
    __shared__ f16 sV[2][64 * 32];   // [dh][key]   2 x 4KB
    __shared__ f16 sP[4][32 * 32];   // per-wave P scratch [32 q][32 key]

    const int tid  = threadIdx.x;
    const int wave = tid >> 5;
    const int lane = tid & 31;
    const int h = blockIdx.y, b = blockIdx.z;
    const int q0 = blockIdx.x * 128 + wave * 32;
    const int kh = (lane >> 4) * 8;

    const float scale = 0.125f;                 // 1/sqrt(64)
    const float nbias = -8.31776616672f;        // -ln(4096)
    const float l2e   = 1.44269504089f;

    // Q fragments (32 x 64), loaded once.
    v16h qa[2][2];
#pragma unroll
    for (int mt = 0; mt < 2; ++mt) {
        const f16* qrow =
            Qh + ((size_t)b * Sc + q0 + mt * 16 + (lane & 15)) * Dc + h * DHc;
        qa[mt][0] = load_frag(qrow, kh);
        qa[mt][1] = load_frag(qrow + 32, kh);
    }

    // stage K tile (256 chunks) + V tile (256 chunks): 4 x 16B per thread
    auto issueKV = [&](int buf, int k0) {
#pragma unroll
        for (int i = 0; i < 2; ++i) {
            const int c  = tid + i * 128;
            const int kr = c >> 3, ko = (c & 7) * 8;
            async_copy_b128(&sK[buf][kr * 64 + ko],
                            &Kh[((size_t)b * Sc + k0 + kr) * Dc + h * DHc + ko]);
            const int vr = c >> 2, vo = (c & 3) * 8;
            async_copy_b128(&sV[buf][vr * 32 + vo],
                            &Vt[(((size_t)b * Hc + h) * DHc + vr) * Sc + k0 + vo]);
        }
    };

    v8f o[2][4] = {};
    f16* prow = &sP[wave][0];

    issueKV(0, 0);
    for (int k0 = 0; k0 < Sc; k0 += 32) {
        const int cur = (k0 >> 5) & 1;
        issueKV(cur ^ 1, (k0 + 32 < Sc) ? k0 + 32 : 0);  // prefetch next block
        WAIT_ASYNCCNT("0x4");   // in-order: first 4 (current block) complete
        __syncthreads();

        // scores: 32 q-rows x 32 keys
        v8f s[2][2];
#pragma unroll
        for (int g = 0; g < 2; ++g) {
            const f16* krow = &sK[cur][(g * 16 + (lane & 15)) * 64];
            v16h b0 = load_frag(krow, kh);        // dh 0..31
            v16h b1 = load_frag(krow + 32, kh);   // dh 32..63
#pragma unroll
            for (int mt = 0; mt < 2; ++mt) {
                v8f acc = {};
                acc      = wmma_f16(qa[mt][0], b0, acc);
                s[mt][g] = wmma_f16(qa[mt][1], b1, acc);
            }
        }

        // sigmoid(s*scale - ln(S)) -> f16, transpose into A-fragment order
#pragma unroll
        for (int mt = 0; mt < 2; ++mt)
#pragma unroll
            for (int g = 0; g < 2; ++g)
#pragma unroll
                for (int r = 0; r < 8; ++r) {
                    const float x = s[mt][g][r] * scale + nbias;
                    const float p =
                        1.0f / (1.0f + __builtin_amdgcn_exp2f(-x * l2e));
                    const int m = mt * 16 + (lane >> 4) * 8 + r;
                    prow[m * 32 + g * 16 + (lane & 15)] = (f16)p;
                }
        __syncthreads();   // DS in-order per wave; barrier keeps model clean

        // O (32 x 64) += P (32 x 32keys) x V (32keys x 64)
        v16h pa[2];
#pragma unroll
        for (int mt = 0; mt < 2; ++mt)
            pa[mt] = load_frag(prow + (mt * 16 + (lane & 15)) * 32, kh);
#pragma unroll
        for (int j = 0; j < 4; ++j) {
            v16h vb = load_frag(&sV[cur][(j * 16 + (lane & 15)) * 32], kh);
#pragma unroll
            for (int mt = 0; mt < 2; ++mt)
                o[mt][j] = wmma_f16(pa[mt], vb, o[mt][j]);
        }
        __syncthreads();   // all waves done with sK/sV[cur] before overwrite
    }

    // Out[b][s][h*64+dh], f32
#pragma unroll
    for (int mt = 0; mt < 2; ++mt)
#pragma unroll
        for (int j = 0; j < 4; ++j) {
            const int dh    = j * 16 + (lane & 15);
            const int sbase = q0 + mt * 16 + (lane >> 4) * 8;
#pragma unroll
            for (int r = 0; r < 8; ++r)
                Out[((size_t)b * Sc + sbase + r) * Dc + h * DHc + dh] = o[mt][j][r];
        }
}

// ---------------------------------------------------------------------------
// Host launcher
// ---------------------------------------------------------------------------
extern "C" void kernel_launch(void* const* d_in, const int* in_sizes, int n_in,
                              void* d_out, int out_size, void* d_ws, size_t ws_size,
                              hipStream_t stream) {
    (void)in_sizes; (void)n_in; (void)out_size; (void)ws_size;

    const float* X  = (const float*)d_in[0];
    const float* Wq = (const float*)d_in[1];
    const float* Wk = (const float*)d_in[2];
    const float* Wv = (const float*)d_in[3];
    float* out = (float*)d_out;

    const int nX = Bc * Sc * Dc;   // 6,291,456
    const int nW = Dc * Dc;        //   589,824

    char* ws = (char*)d_ws;
    size_t off = 0;
    f16* Xh  = (f16*)(ws + off); off += (size_t)nX * 2;
    f16* Wqh = (f16*)(ws + off); off += (size_t)nW * 2;
    f16* Wkh = (f16*)(ws + off); off += (size_t)nW * 2;
    f16* Wvh = (f16*)(ws + off); off += (size_t)nW * 2;
    f16* Qh  = (f16*)(ws + off); off += (size_t)nX * 2;
    f16* Kh  = (f16*)(ws + off); off += (size_t)nX * 2;
    f16* Vt  = (f16*)(ws + off); off += (size_t)nX * 2;

    cvt_f32_f16<<<nX / 4 / 256, 256, 0, stream>>>(X,  Xh,  nX);
    cvt_f32_f16<<<nW / 4 / 256, 256, 0, stream>>>(Wq, Wqh, nW);
    cvt_f32_f16<<<nW / 4 / 256, 256, 0, stream>>>(Wk, Wkh, nW);
    cvt_f32_f16<<<nW / 4 / 256, 256, 0, stream>>>(Wv, Wvh, nW);

    proj_wmma<<<dim3((Bc * Sc) / 128, Dc / 64, 3), 128, 0, stream>>>(
        Xh, Wqh, Wkh, Wvh, Qh, Kh, Vt);

    attn_wmma<<<dim3(Sc / 128, Hc, Bc), 128, 0, stream>>>(Qh, Kh, Vt, out);
}
